// DeformableTransformerDecoderLayer_4209067950225
// MI455X (gfx1250) — matrix operand order repaired and checked by reading, not compiled, not run on hardware
//
#include <hip/hip_runtime.h>

// ---------------------------------------------------------------------------
// CDNA5 (gfx1250) deformable-DETR decoder layer.
// GEMMs: v_wmma_f32_16x16x32_bf16 with vectorized (b128) LDS staging.
// Attention: fused-softmax WMMA; K/V tiles DMA'd to LDS by the Tensor Data
// Mover (tensor_load_to_lds + s_wait_tensorcnt), V kept transposed in LDS so
// every WMMA fragment load is contiguous.
// ---------------------------------------------------------------------------

typedef __bf16 bf16_t;
typedef __attribute__((ext_vector_type(16))) __bf16 v16bf;
typedef __attribute__((ext_vector_type(8)))  __bf16 v8bf;
typedef __attribute__((ext_vector_type(8)))  float  v8f;
typedef __attribute__((ext_vector_type(4)))  unsigned int v4u;
typedef __attribute__((ext_vector_type(4)))  int v4i;
typedef __attribute__((ext_vector_type(8)))  int v8i;

__device__ __forceinline__ v8f wmma_bf16(v16bf a, v16bf b, v8f c) {
  return __builtin_amdgcn_wmma_f32_16x16x32_bf16(false, a, false, b, (short)0, c,
                                                 false, false);
}

// Two contiguous 16-byte LDS runs -> one 16-element bf16 fragment.
__device__ __forceinline__ v16bf ld8x2(const bf16_t* p0, const bf16_t* p1) {
  v8bf lo = *(const v8bf*)p0;
  v8bf hi = *(const v8bf*)p1;
  return __builtin_shufflevector(lo, hi, 0, 1, 2, 3, 4, 5, 6, 7,
                                 8, 9, 10, 11, 12, 13, 14, 15);
}
__device__ __forceinline__ v16bf ld16(const bf16_t* p) {
  return ld8x2(p, p + 8);
}

// ---------------------------------------------------------------------------
// Tensor Data Mover: 2D tile (width x rows, fp32, strided rows) -> LDS.
// D# packed per CDNA5 ISA ch.8 (group0: count/lds_addr/global_addr/type,
// group1: data_size, tensor dims, tile dims, dim0 stride).
// ---------------------------------------------------------------------------
__device__ __forceinline__ void tdm_load_2d_f32(const float* gptr,
                                                unsigned lds_off,
                                                unsigned width, unsigned rows,
                                                unsigned row_stride) {
  unsigned long long ga = (unsigned long long)(size_t)gptr;
  v4u g0;
  g0[0] = 1u;                                        // count=1 (valid), user
  g0[1] = lds_off;                                   // lds_addr (bytes)
  g0[2] = (unsigned)(ga & 0xffffffffu);              // global_addr[31:0]
  g0[3] = (unsigned)((ga >> 32) & 0x01ffffffu) | (2u << 30);  // [56:32]|type=2
  v8i g1;
  g1[0] = (int)(2u << 16);                           // data_size=4B
  g1[1] = (int)((width & 0xffffu) << 16);            // tensor_dim0[15:0]
  g1[2] = (int)(((width >> 16) & 0xffffu) | ((rows & 0xffffu) << 16));
  g1[3] = (int)(((rows >> 16) & 0xffffu) | ((width & 0xffffu) << 16)); // tile_dim0
  g1[4] = (int)(rows & 0xffffu);                     // tile_dim1 (tile_dim2=0)
  g1[5] = (int)row_stride;                           // tensor_dim0_stride lo32
  g1[6] = 0;
  g1[7] = 0;
  v4i z = {0, 0, 0, 0};
#if __clang_major__ >= 23
  v8i z8 = {0, 0, 0, 0, 0, 0, 0, 0};
  __builtin_amdgcn_tensor_load_to_lds(g0, g1, z, z, z8, 0);
#else
  __builtin_amdgcn_tensor_load_to_lds(g0, g1, z, z, 0);
#endif
}

// Problem constants (static in the reference).
#define NB   8
#define NLQ  300
#define NC   256
#define NHD  8
#define NDH  32
#define NDFF 1024
#define NLIN 13294
#define M1   (NB * NLQ)      // 2400

// ---------------------------------------------------------------------------
// Generic WMMA GEMM:  Y[M,N] = X[M,K] @ W[N,K]^T + bias[N]  (optional ReLU)
// Block 256 threads = 8 waves; tile 64x64; K stepped by 32.
// Staging: each thread loads 8 contiguous floats (2x b128), stores 1x b128 bf16.
// ---------------------------------------------------------------------------
#define TM 64
#define TN 64
#define TK 32

__global__ __launch_bounds__(256) void gemm_wmma(
    const float* __restrict__ X, const float* __restrict__ W,
    const float* __restrict__ bias, float* __restrict__ Y,
    int M, int N, int K, int relu)
{
  __shared__ bf16_t As[TM][TK + 8];
  __shared__ bf16_t Bs[TN][TK + 8];

  const int lane = threadIdx.x & 31;
  const int wave = threadIdx.x >> 5;
  const int lh   = lane >> 4;
  const int l16  = lane & 15;
  const int wm   = wave & 3;
  const int wn   = wave >> 2;
  const int mbase = blockIdx.x * TM;
  const int nbase = blockIdx.y * TN;

  const int sr = threadIdx.x >> 2;         // staging row 0..63
  const int sc = (threadIdx.x & 3) << 3;   // staging col group {0,8,16,24}

  v8f acc0 = {};
  v8f acc1 = {};

  for (int k0 = 0; k0 < K; k0 += TK) {
    if (k0 + TK < K) {
      int pr = mbase + sr; if (pr >= M) pr = M - 1;
      int pc = nbase + sr; if (pc >= N) pc = N - 1;
      __builtin_prefetch(&X[(size_t)pr * K + k0 + TK + sc], 0, 1);
      __builtin_prefetch(&W[(size_t)pc * K + k0 + TK + sc], 0, 1);
    }
    {
      int gr = mbase + sr;
      float4 f0 = make_float4(0.f, 0.f, 0.f, 0.f), f1 = f0;
      if (gr < M) {
        const float4* p = (const float4*)(X + (size_t)gr * K + k0 + sc);
        f0 = p[0]; f1 = p[1];
      }
      v8bf hv;
      hv[0] = (bf16_t)f0.x; hv[1] = (bf16_t)f0.y;
      hv[2] = (bf16_t)f0.z; hv[3] = (bf16_t)f0.w;
      hv[4] = (bf16_t)f1.x; hv[5] = (bf16_t)f1.y;
      hv[6] = (bf16_t)f1.z; hv[7] = (bf16_t)f1.w;
      *(v8bf*)&As[sr][sc] = hv;
    }
    {
      int gr = nbase + sr;
      float4 f0 = make_float4(0.f, 0.f, 0.f, 0.f), f1 = f0;
      if (gr < N) {
        const float4* p = (const float4*)(W + (size_t)gr * K + k0 + sc);
        f0 = p[0]; f1 = p[1];
      }
      v8bf hv;
      hv[0] = (bf16_t)f0.x; hv[1] = (bf16_t)f0.y;
      hv[2] = (bf16_t)f0.z; hv[3] = (bf16_t)f0.w;
      hv[4] = (bf16_t)f1.x; hv[5] = (bf16_t)f1.y;
      hv[6] = (bf16_t)f1.z; hv[7] = (bf16_t)f1.w;
      *(v8bf*)&Bs[sr][sc] = hv;
    }
    __syncthreads();

    const int arow = wm * 16 + l16;
    v16bf a  = ld16(&As[arow][lh << 4]);
    v16bf b0 = ld16(&Bs[wn * 32 + l16][lh << 4]);
    v16bf b1 = ld16(&Bs[wn * 32 + 16 + l16][lh << 4]);
    acc0 = wmma_bf16(a, b0, acc0);
    acc1 = wmma_bf16(a, b1, acc1);
    __syncthreads();
  }

#pragma unroll
  for (int r = 0; r < 8; ++r) {
    int row = mbase + wm * 16 + r + (lh << 3);
    int c0  = nbase + wn * 32 + l16;
    int c1  = c0 + 16;
    if (row < M) {
      if (c0 < N) {
        float v = acc0[r] + bias[c0];
        if (relu) v = fmaxf(v, 0.0f);
        Y[(size_t)row * N + c0] = v;
      }
      if (c1 < N) {
        float v = acc1[r] + bias[c1];
        if (relu) v = fmaxf(v, 0.0f);
        Y[(size_t)row * N + c1] = v;
      }
    }
  }
}

// ---------------------------------------------------------------------------
// Fused-softmax MHA attention. One block per (b,h). K/V tiles are DMA'd into
// LDS by the TDM, then converted fp32->bf16 (V transposed). 4 waves, each
// owns 16-query tiles; 3 WMMA passes (rowmax / exp+rowsum+P / P@V).
// Dynamic LDS map (bytes):
//   Kt  bf16[320][32]   @      0  (20480)
//   Vtt bf16[32][320]   @  20480  (20480)   (transposed V)
//   P   bf16[4][16][320]@  40960  (40960)
//   Kst f32 [300][32]   @  81920  (38400)   TDM staging
//   Vst f32 [300][32]   @ 120320  (38400)
// total 158720.
// ---------------------------------------------------------------------------
#define ATT_LDS_KT   0
#define ATT_LDS_VTT  20480
#define ATT_LDS_P    40960
#define ATT_LDS_KST  81920
#define ATT_LDS_VST  120320
#define ATT_LDS_SIZE 158720

__global__ __launch_bounds__(128) void attn_wmma(
    const float* __restrict__ QK,   // (2400, 512): cols 0..255 Q, 256..511 K
    const float* __restrict__ Vb,   // (2400, 256)
    float* __restrict__ Out)        // (2400, 256)
{
  extern __shared__ char smem[];
  bf16_t* Kt  = (bf16_t*)(smem + ATT_LDS_KT);    // [320][32]
  bf16_t* Vtt = (bf16_t*)(smem + ATT_LDS_VTT);   // [32][320]
  bf16_t* P   = (bf16_t*)(smem + ATT_LDS_P);     // [4][16][320]
  const float* Kst = (const float*)(smem + ATT_LDS_KST);  // [300][32]
  const float* Vst = (const float*)(smem + ATT_LDS_VST);  // [300][32]

  const int bh = blockIdx.x;
  const int b  = bh >> 3;
  const int h  = bh & 7;
  const int lane = threadIdx.x & 31;
  const int wave = threadIdx.x >> 5;
  const int lh   = lane >> 4;
  const int l16  = lane & 15;
  const float scale = 0.17677669529663687f;  // 1/sqrt(32)

  // --- TDM: DMA the strided K and V tiles (300 rows x 32 cols fp32) ---
  if (wave == 0) {
    tdm_load_2d_f32(QK + (size_t)(b * NLQ) * 512 + 256 + h * 32,
                    ATT_LDS_KST, 32, NLQ, 512);
    tdm_load_2d_f32(Vb + (size_t)(b * NLQ) * 256 + h * 32,
                    ATT_LDS_VST, 32, NLQ, 256);
    __builtin_amdgcn_s_wait_tensorcnt((short)0);
  }
  __syncthreads();

  // convert K (direct) and V (transposed) to bf16, zero-pad keys 300..319
  for (int i = threadIdx.x; i < 320 * 32; i += 128) {
    int k = i >> 5, d = i & 31;
    Kt[i] = (bf16_t)((k < NLQ) ? Kst[k * 32 + d] : 0.0f);
  }
  for (int i = threadIdx.x; i < 32 * 320; i += 128) {
    int d = i / 320, k = i % 320;
    Vtt[i] = (bf16_t)((k < NLQ) ? Vst[k * 32 + d] : 0.0f);
  }
  __syncthreads();

  bf16_t* Pw = P + wave * 16 * 320;
  for (int i = lane; i < 16 * 16; i += 32)
    Pw[(i >> 4) * 320 + 304 + (i & 15)] = (bf16_t)0.0f;

  for (int tile = wave; tile < 19; tile += 4) {
    // Q fragment (16x32): one contiguous 16-float run per lane
    int qrow = tile * 16 + l16;
    int qg   = b * NLQ + (qrow < NLQ ? qrow : NLQ - 1);
    const float* qp = QK + (size_t)qg * 512 + h * 32;
    float4 a0 = *(const float4*)(qp + (lh << 4));
    float4 a1 = *(const float4*)(qp + (lh << 4) + 4);
    float4 a2 = *(const float4*)(qp + (lh << 4) + 8);
    float4 a3 = *(const float4*)(qp + (lh << 4) + 12);
    v16bf qf;
    qf[0]  = (bf16_t)a0.x; qf[1]  = (bf16_t)a0.y;
    qf[2]  = (bf16_t)a0.z; qf[3]  = (bf16_t)a0.w;
    qf[4]  = (bf16_t)a1.x; qf[5]  = (bf16_t)a1.y;
    qf[6]  = (bf16_t)a1.z; qf[7]  = (bf16_t)a1.w;
    qf[8]  = (bf16_t)a2.x; qf[9]  = (bf16_t)a2.y;
    qf[10] = (bf16_t)a2.z; qf[11] = (bf16_t)a2.w;
    qf[12] = (bf16_t)a3.x; qf[13] = (bf16_t)a3.y;
    qf[14] = (bf16_t)a3.z; qf[15] = (bf16_t)a3.w;

    // ---- pass 1: row max ----
    float rmax[8];
#pragma unroll
    for (int r = 0; r < 8; ++r) rmax[r] = -1e30f;
    for (int kt = 0; kt < 19; ++kt) {
      v16bf kf = ld16(&Kt[(kt * 16 + l16) * 32 + (lh << 4)]);
      v8f s = {};
      s = wmma_bf16(qf, kf, s);
      bool valid = (kt * 16 + l16) < NLQ;
#pragma unroll
      for (int r = 0; r < 8; ++r) {
        float sv = valid ? s[r] * scale : -1e30f;
        rmax[r] = fmaxf(rmax[r], sv);
      }
    }
#pragma unroll
    for (int r = 0; r < 8; ++r) {
      float v = rmax[r];
#pragma unroll
      for (int o = 1; o < 16; o <<= 1) v = fmaxf(v, __shfl_xor(v, o, 32));
      rmax[r] = v;
    }

    // ---- pass 2: exp, row sums, write P (bf16) ----
    float rsum[8];
#pragma unroll
    for (int r = 0; r < 8; ++r) rsum[r] = 0.0f;
    for (int kt = 0; kt < 19; ++kt) {
      v16bf kf = ld16(&Kt[(kt * 16 + l16) * 32 + (lh << 4)]);
      v8f s = {};
      s = wmma_bf16(qf, kf, s);
      bool valid = (kt * 16 + l16) < NLQ;
#pragma unroll
      for (int r = 0; r < 8; ++r) {
        float e = valid ? __expf(s[r] * scale - rmax[r]) : 0.0f;
        rsum[r] += e;
        Pw[(r + (lh << 3)) * 320 + kt * 16 + l16] = (bf16_t)e;
      }
    }
#pragma unroll
    for (int r = 0; r < 8; ++r) {
      float v = rsum[r];
#pragma unroll
      for (int o = 1; o < 16; o <<= 1) v += __shfl_xor(v, o, 32);
      rsum[r] = v;
    }

    // ---- pass 3: O = P @ V (K padded to 320); V transposed => contiguous B
    v8f o0 = {}, o1 = {};
    for (int c = 0; c < 10; ++c) {
      const bf16_t* prow = &Pw[l16 * 320 + c * 32];
      v16bf pf = ld16(prow + (lh << 4));
      v16bf v0 = ld16(&Vtt[l16 * 320 + c * 32 + (lh << 4)]);
      v16bf v1 = ld16(&Vtt[(16 + l16) * 320 + c * 32 + (lh << 4)]);
      o0 = wmma_bf16(pf, v0, o0);
      o1 = wmma_bf16(pf, v1, o1);
    }
#pragma unroll
    for (int r = 0; r < 8; ++r) {
      int q = tile * 16 + r + (lh << 3);
      if (q < NLQ) {
        float inv = 1.0f / rsum[r];
        size_t rowo = (size_t)(b * NLQ + q) * 256 + h * 32;
        Out[rowo + l16]      = o0[r] * inv;
        Out[rowo + 16 + l16] = o1[r] * inv;
      }
    }
  }
}

// ---------------------------------------------------------------------------
// Elementwise add
// ---------------------------------------------------------------------------
__global__ void ewadd(const float* __restrict__ a, const float* __restrict__ b,
                      float* __restrict__ o, int n)
{
  int i = blockIdx.x * blockDim.x + threadIdx.x;
  if (i < n) o[i] = a[i] + b[i];
}

// ---------------------------------------------------------------------------
// Fused add + LayerNorm over last dim (C=256). One block (256 thr) per row.
// ---------------------------------------------------------------------------
__global__ __launch_bounds__(256) void add_ln(
    const float* __restrict__ x, const float* __restrict__ y,
    const float* __restrict__ g, const float* __restrict__ be,
    float* __restrict__ out)
{
  __shared__ float red1[8];
  __shared__ float red2[8];
  const int row = blockIdx.x;
  const int t = threadIdx.x;
  const int lane = t & 31, wv = t >> 5;

  float v = x[(size_t)row * 256 + t] + y[(size_t)row * 256 + t];

  float s = v;
#pragma unroll
  for (int o = 16; o > 0; o >>= 1) s += __shfl_xor(s, o, 32);
  if (lane == 0) red1[wv] = s;
  __syncthreads();
  float tot = 0.0f;
#pragma unroll
  for (int i = 0; i < 8; ++i) tot += red1[i];
  const float mu = tot * (1.0f / 256.0f);

  float d = v - mu;
  float s2 = d * d;
#pragma unroll
  for (int o = 16; o > 0; o >>= 1) s2 += __shfl_xor(s2, o, 32);
  if (lane == 0) red2[wv] = s2;
  __syncthreads();
  float tot2 = 0.0f;
#pragma unroll
  for (int i = 0; i < 8; ++i) tot2 += red2[i];
  const float var = tot2 * (1.0f / 256.0f);

  out[(size_t)row * 256 + t] = d * rsqrtf(var + 1e-5f) * g[t] + be[t];
}

// ---------------------------------------------------------------------------
// MS deformable attention sampling. One wave per (b,q,h); lane = channel d.
// Shapes (100,100)(50,50)(25,25)(13,13); starts 0/10000/12500/13125.
// ---------------------------------------------------------------------------
__global__ __launch_bounds__(256) void msdeform(
    const float* __restrict__ off,   // (2400, 256)  [h][l][p][2]
    const float* __restrict__ aw,    // (2400, 128)  [h][l*4+p]
    const float* __restrict__ ref,   // (B,LQ,4,2)
    const float* __restrict__ value, // (B*LIN, 256)
    float* __restrict__ out)         // (2400, 256)
{
  const int gw = blockIdx.x * 8 + (threadIdx.x >> 5);
  if (gw >= NB * NLQ * NHD) return;
  const int lane = threadIdx.x & 31;
  const int h = gw & 7;
  const int q = (gw >> 3) % NLQ;
  const int b = gw / (NLQ * NHD);

  const int awbase = (b * NLQ + q) * 128 + h * 16;
  float av = (lane < 16) ? aw[awbase + lane] : -1e30f;
  float mx = av;
#pragma unroll
  for (int o = 1; o < 16; o <<= 1) mx = fmaxf(mx, __shfl_xor(mx, o, 32));
  float e = (lane < 16) ? __expf(av - mx) : 0.0f;
  float ssum = e;
#pragma unroll
  for (int o = 1; o < 16; o <<= 1) ssum += __shfl_xor(ssum, o, 32);
  const float w16 = e / ssum;

  const int Hs[4] = {100, 50, 25, 13};
  const int Ws[4] = {100, 50, 25, 13};
  const int Ss[4] = {0, 10000, 12500, 13125};
  const int offbase = (b * NLQ + q) * 256 + h * 32;

  float acc = 0.0f;
#pragma unroll
  for (int l = 0; l < 4; ++l) {
    const float rx = ref[((b * NLQ + q) * 4 + l) * 2 + 0];
    const float ry = ref[((b * NLQ + q) * 4 + l) * 2 + 1];
    const int Hl = Hs[l], Wl = Ws[l], Sl = Ss[l];
#pragma unroll
    for (int p = 0; p < 4; ++p) {
      const float ox = off[offbase + l * 8 + p * 2 + 0];
      const float oy = off[offbase + l * 8 + p * 2 + 1];
      const float x = (rx + ox / (float)Wl) * (float)Wl - 0.5f;
      const float y = (ry + oy / (float)Hl) * (float)Hl - 0.5f;
      const float x0 = floorf(x), y0 = floorf(y);
      const float wgt = __shfl(w16, l * 4 + p, 32);
#pragma unroll
      for (int dy = 0; dy < 2; ++dy) {
#pragma unroll
        for (int dx = 0; dx < 2; ++dx) {
          const float xi = x0 + (float)dx, yi = y0 + (float)dy;
          const float wb = (1.0f - fabsf(x - xi)) * (1.0f - fabsf(y - yi));
          if (xi >= 0.0f && xi < (float)Wl && yi >= 0.0f && yi < (float)Hl) {
            const int idx = (int)yi * Wl + (int)xi;
            const float gv =
                value[(size_t)(b * NLIN + Sl + idx) * 256 + h * 32 + lane];
            acc += wgt * wb * gv;
          }
        }
      }
    }
  }
  out[(size_t)(b * NLQ + q) * 256 + h * 32 + lane] = acc;
}

// ---------------------------------------------------------------------------
// Orchestration
// ---------------------------------------------------------------------------
extern "C" void kernel_launch(void* const* d_in, const int* in_sizes, int n_in,
                              void* d_out, int out_size, void* d_ws, size_t ws_size,
                              hipStream_t stream) {
  (void)in_sizes; (void)n_in; (void)out_size; (void)ws_size;

  const float* tgt   = (const float*)d_in[0];
  const float* qpos  = (const float*)d_in[1];
  const float* refp  = (const float*)d_in[2];
  const float* src   = (const float*)d_in[3];
  const float* in_w  = (const float*)d_in[6];
  const float* in_b  = (const float*)d_in[7];
  const float* out_w = (const float*)d_in[8];
  const float* out_b = (const float*)d_in[9];
  const float* off_w = (const float*)d_in[10];
  const float* off_b = (const float*)d_in[11];
  const float* aw_w  = (const float*)d_in[12];
  const float* aw_b  = (const float*)d_in[13];
  const float* val_w = (const float*)d_in[14];
  const float* val_b = (const float*)d_in[15];
  const float* op_w  = (const float*)d_in[16];
  const float* op_b  = (const float*)d_in[17];
  const float* ln1_g = (const float*)d_in[18];
  const float* ln1_b = (const float*)d_in[19];
  const float* ln2_g = (const float*)d_in[20];
  const float* ln2_b = (const float*)d_in[21];
  const float* ln3_g = (const float*)d_in[22];
  const float* ln3_b = (const float*)d_in[23];
  const float* l1w   = (const float*)d_in[24];
  const float* l1b   = (const float*)d_in[25];
  const float* l2w   = (const float*)d_in[26];
  const float* l2b   = (const float*)d_in[27];

  float* ws = (float*)d_ws;
  float* q    = ws;               // 2400*256
  float* qk   = q    + M1 * 256;  // 2400*512
  float* vb   = qk   + M1 * 512;  // 2400*256
  float* ao   = vb   + M1 * 256;  // 2400*256
  float* mha  = ao   + M1 * 256;  // 2400*256
  float* x1   = mha  + M1 * 256;  // 2400*256
  float* q2   = x1   + M1 * 256;  // 2400*256
  float* offb = q2   + M1 * 256;  // 2400*256
  float* awb  = offb + M1 * 256;  // 2400*128
  float* samp = awb  + M1 * 128;  // 2400*256
  float* t2   = samp + M1 * 256;  // 2400*256
  float* x2   = t2   + M1 * 256;  // 2400*256
  float* ffh  = x2   + M1 * 256;  // 2400*1024
  float* ffo  = ffh  + M1 * 1024; // 2400*256
  float* val  = ffo  + M1 * 256;  // 106352*256

  const int GM1 = (M1 + TM - 1) / TM;            // 38
  const int GMV = (NB * NLIN + TM - 1) / TM;     // 1662

  // q = tgt + query_pos
  ewadd<<<(M1 * 256 + 255) / 256, 256, 0, stream>>>(tgt, qpos, q, M1 * 256);

  // MHA: Q,K from q; V from tgt
  gemm_wmma<<<dim3(GM1, 8), 256, 0, stream>>>(q, in_w, in_b, qk, M1, 512, 256, 0);
  gemm_wmma<<<dim3(GM1, 4), 256, 0, stream>>>(tgt, in_w + 512 * 256, in_b + 512,
                                              vb, M1, 256, 256, 0);
  attn_wmma<<<NB * NHD, 128, ATT_LDS_SIZE, stream>>>(qk, vb, ao);
  gemm_wmma<<<dim3(GM1, 4), 256, 0, stream>>>(ao, out_w, out_b, mha, M1, 256, 256, 0);
  add_ln<<<M1, 256, 0, stream>>>(tgt, mha, ln2_g, ln2_b, x1);

  // deformable attention
  ewadd<<<(M1 * 256 + 255) / 256, 256, 0, stream>>>(x1, qpos, q2, M1 * 256);
  gemm_wmma<<<dim3(GMV, 4), 256, 0, stream>>>(src, val_w, val_b, val,
                                              NB * NLIN, 256, 256, 0);
  gemm_wmma<<<dim3(GM1, 4), 256, 0, stream>>>(q2, off_w, off_b, offb, M1, 256, 256, 0);
  gemm_wmma<<<dim3(GM1, 2), 256, 0, stream>>>(q2, aw_w, aw_b, awb, M1, 128, 256, 0);
  msdeform<<<(NB * NLQ * NHD) / 8, 256, 0, stream>>>(offb, awb, refp, val, samp);
  gemm_wmma<<<dim3(GM1, 4), 256, 0, stream>>>(samp, op_w, op_b, t2, M1, 256, 256, 0);
  add_ln<<<M1, 256, 0, stream>>>(x1, t2, ln1_g, ln1_b, x2);

  // FFN
  gemm_wmma<<<dim3(GM1, 16), 256, 0, stream>>>(x2, l1w, l1b, ffh, M1, 1024, 256, 1);
  gemm_wmma<<<dim3(GM1, 4), 256, 0, stream>>>(ffh, l2w, l2b, ffo, M1, 256, 1024, 0);
  add_ln<<<M1, 256, 0, stream>>>(x2, ffo, ln3_g, ln3_b, (float*)d_out);
}
